// DeepWF_18820546691332
// MI455X (gfx1250) — matrix operand order repaired and checked by reading, not compiled, not run on hardware
//
#include <hip/hip_runtime.h>
#include <hip/hip_bf16.h>
#include <math.h>

typedef __bf16 bf16;
typedef __attribute__((ext_vector_type(16))) __bf16 v16bf;
typedef __attribute__((ext_vector_type(8)))  __bf16 v8bf;
typedef __attribute__((ext_vector_type(8)))  float  v8f;

#define SILU_GAIN 1.7868129431578026f
#define TANH_GAIN 1.5927812698663606f
#define RSQRT2    0.70710678118654752f

__device__ __forceinline__ float silu_g(float x){ return SILU_GAIN * x * (1.0f/(1.0f+__expf(-x))); }
__device__ __forceinline__ float tanh_g(float x){ return TANH_GAIN * tanhf(x); }

// ---------------------------------------------------------------------------
// WMMA bf16 GEMM, register-blocked along N:
//   each wave computes a 16 x (16*NT) C strip with NT independent accumulator
//   chains (fills the XDL pipeline; A fragment loaded once per K-step).
// C[M,N] = act(alpha*(A[M,K] @ B[K,N] + bias[N]));  B supplied as Bt[N,K].
// Optional residual: C = (resid + act(...)) * RSQRT2 (same-element, in-place ok)
// act: 0=none, 1=silu*gain, 2=tanh*gain.  M%16==0, K%32==0, (N/16)%NT==0.
// ---------------------------------------------------------------------------
template<int NT>
__global__ void gemm_bf16_wmma(const bf16* __restrict__ A, const bf16* __restrict__ Bt,
                               const float* __restrict__ bias, const float* __restrict__ resid,
                               float* __restrict__ C, int M, int K, int N,
                               float alpha, int act)
{
    const int mtiles = M >> 4, ngroups = (N >> 4) / NT;
    int wave = blockIdx.x * (blockDim.x >> 5) + (threadIdx.x >> 5);
    if (wave >= mtiles * ngroups) return;
    int mt = wave / ngroups, ng = wave % ngroups;
    int lane = threadIdx.x & 31;
    int l15 = lane & 15, hi = lane >> 4;

    // A frag (16-bit A 16x32): lane half 'hi' holds K { hi*16 .. hi*16+15 }
    const bf16* arow = A + (size_t)(mt*16 + l15) * K + hi*16;
    // B frag (16-bit B 32x16): lane = column, lane half holds K { hi*16 .. hi*16+15 }
    const bf16* brow[NT];
#pragma unroll
    for (int t = 0; t < NT; ++t)
        brow[t] = Bt + (size_t)((ng*NT + t)*16 + l15) * K + hi*16;

    v8f c[NT] = {};
    for (int k = 0; k < K; k += 32) {
        v8bf a0 = *(const v8bf*)(arow + k);
        v8bf a1 = *(const v8bf*)(arow + k + 8);
        v16bf a = __builtin_shufflevector(a0, a1, 0,1,2,3,4,5,6,7,8,9,10,11,12,13,14,15);
#pragma unroll
        for (int t = 0; t < NT; ++t) {
            v16bf b = *(const v16bf*)(brow[t] + k);
            c[t] = __builtin_amdgcn_wmma_f32_16x16x32_bf16(false, a, false, b, (short)0, c[t], false, false);
        }
    }

#pragma unroll
    for (int t = 0; t < NT; ++t) {
        int col = (ng*NT + t)*16 + l15;
        float bv = bias ? bias[col] : 0.0f;
#pragma unroll
        for (int r = 0; r < 8; ++r) {
            int row = mt*16 + r + hi*8;          // C layout: lanes16-31 hold M = r+8
            float v = (c[t][r] + bv) * alpha;
            if (act == 1)      v = silu_g(v);
            else if (act == 2) v = tanh_g(v);
            size_t idx = (size_t)row * N + col;
            if (resid) v = (resid[idx] + v) * RSQRT2;
            C[idx] = v;
        }
    }
}

// --------------------------- small helper kernels ---------------------------

__global__ void f32_to_bf16_k(const float* __restrict__ in, bf16* __restrict__ out, int n){
    int i = blockIdx.x*blockDim.x + threadIdx.x;
    if (i < n) out[i] = (bf16)in[i];
}

// Bt[n*Kpad + k] = (k<K) ? W[k*N+n] : 0   (transpose + convert + zero-pad K)
__global__ void tconv_k(const float* __restrict__ W, bf16* __restrict__ Bt, int K, int N, int Kpad){
    int idx = blockIdx.x*blockDim.x + threadIdx.x;
    if (idx >= N*Kpad) return;
    int n = idx / Kpad, k = idx % Kpad;
    Bt[idx] = (k < K) ? (bf16)W[k*N + n] : (bf16)0.0f;
}

// InvariantEncoding: per-electron, mean over 32 atoms of tanh-MLP(32->32->32)
__global__ void encode_kernel(const float* __restrict__ eW, const float* __restrict__ eb,
                              const float* __restrict__ nuc,
                              const float* __restrict__ m1W, const float* __restrict__ m1b,
                              const float* __restrict__ m2W, const float* __restrict__ m2b,
                              const float* __restrict__ el, const float* __restrict__ at,
                              float* __restrict__ hone0)
{
    int i = blockIdx.x, c = threadIdx.x;      // 512 blocks x 32 threads
    __shared__ float hs[32], ts[32];
    float ex = el[i*3+0], ey = el[i*3+1], ez = el[i*3+2];
    float acc = 0.f;
    for (int m = 0; m < 32; ++m) {
        float rx = ex - at[m*3+0], ry = ey - at[m*3+1], rz = ez - at[m*3+2];
        float rn = sqrtf(rx*rx + ry*ry + rz*rz);
        float h = eb[c] + rx*eW[0*32+c] + ry*eW[1*32+c] + rz*eW[2*32+c] + rn*eW[3*32+c];
        h = (h + nuc[m*32+c]) * RSQRT2;
        hs[c] = h; __syncthreads();
        float t = m1b[c];
        for (int k = 0; k < 32; ++k) t += hs[k]*m1W[k*32+c];
        ts[c] = tanh_g(t); __syncthreads();
        float u = m2b[c];
        for (int k = 0; k < 32; ++k) u += ts[k]*m2W[k*32+c];
        acc += tanh_g(u);
        __syncthreads();
    }
    hone0[i*32+c] = acc * (1.0f/32.0f);
}

// h_two init: (rx,ry,rz, i==j?0:|r|) stored p-fastest, p=4
__global__ void htwo_init(const float* __restrict__ el, float* __restrict__ ht0){
    int idx = blockIdx.x*blockDim.x + threadIdx.x;
    if (idx >= 512*512) return;
    int i = idx >> 9, j = idx & 511;
    float rx = el[i*3+0]-el[j*3+0], ry = el[i*3+1]-el[j*3+1], rz = el[i*3+2]-el[j*3+2];
    float rn = (i==j) ? 0.f : sqrtf(rx*rx+ry*ry+rz*rz);
    float* o = ht0 + (size_t)idx*4;
    o[0]=rx; o[1]=ry; o[2]=rz; o[3]=rn;
}

// spin means over first index of h_two[i][j][c]
__global__ void pair_mean(const float* __restrict__ ht, int p,
                          float* __restrict__ up, float* __restrict__ dn){
    int idx = blockIdx.x*blockDim.x + threadIdx.x;
    if (idx >= 512*p) return;
    int j = idx / p, c = idx % p;
    float su = 0.f, sd = 0.f;
    for (int i = 0;   i < 256; ++i) su += ht[((size_t)i*512 + j)*p + c];
    for (int i = 256; i < 512; ++i) sd += ht[((size_t)i*512 + j)*p + c];
    up[idx] = su*(1.f/256.f); dn[idx] = sd*(1.f/256.f);
}

// rowbias[c] = bo[c] + (g @ Wg)[c], g = [mean_up h_one | mean_dn h_one] (2s)
__global__ void gmean_rowbias(const float* __restrict__ hone, int s,
                              const float* __restrict__ Wg, const float* __restrict__ bo,
                              float* __restrict__ rowbias){
    __shared__ float g[512];
    int t = threadIdx.x; // 256
    for (int c = t; c < 2*s; c += 256) {
        int ch = c % s; int base = (c < s) ? 0 : 256;
        float sum = 0.f;
        for (int i = 0; i < 256; ++i) sum += hone[(size_t)(base+i)*s + ch];
        g[c] = sum*(1.f/256.f);
    }
    __syncthreads();
    float rb = bo[t];
    for (int k = 0; k < 2*s; ++k) rb += g[k]*Wg[k*256 + t];
    rowbias[t] = rb;
}

// one_in = [h_one | up_mean | dn_mean | 0-pad] -> bf16 (512 x Kpad)
__global__ void assemble_onein(const float* __restrict__ hone, int s,
                               const float* __restrict__ up, const float* __restrict__ dn, int p,
                               bf16* __restrict__ out, int Kpad){
    int idx = blockIdx.x*blockDim.x + threadIdx.x;
    if (idx >= 512*Kpad) return;
    int i = idx / Kpad, c = idx % Kpad;
    float v;
    if      (c < s)       v = hone[(size_t)i*s + c];
    else if (c < s + p)   v = up[i*p + (c - s)];
    else if (c < s + 2*p) v = dn[i*p + (c - s - p)];
    else                  v = 0.f;
    out[idx] = (bf16)v;
}

// layer-0 pair update (K=4, memory-bound -> VALU)
__global__ void pair0_kernel(const float* __restrict__ ht0, const float* __restrict__ Wp,
                             const float* __restrict__ bp, float* __restrict__ ht){
    int idx = blockIdx.x*blockDim.x + threadIdx.x;
    if (idx >= 262144*32) return;
    int r = idx >> 5, c = idx & 31;
    const float* x = ht0 + (size_t)r*4;
    float v = bp[c] + x[0]*Wp[c] + x[1]*Wp[32+c] + x[2]*Wp[64+c] + x[3]*Wp[96+c];
    ht[idx] = silu_g(v);
}

// triu(k=1) pair table, row-major rank (matches np.triu_indices)
__global__ void pair_idx_k(int* __restrict__ pairs){
    int idx = blockIdx.x*blockDim.x + threadIdx.x;
    if (idx >= 65536) return;
    int i = idx >> 8, j = idx & 255;
    if (j <= i) return;
    int p = i*(511 - i)/2 + (j - i - 1);
    pairs[p] = i | (j << 16);
}

// HA = h @ W1[:256], HB = h @ W1[256:]  (h: 256x256 block of h_one)
__global__ void hab_kernel(const float* __restrict__ h, const float* __restrict__ W1,
                           float* __restrict__ HA, float* __restrict__ HB){
    __shared__ float hrow[256];
    int r = blockIdx.x, t = threadIdx.x;
    hrow[t] = h[(size_t)r*256 + t];
    __syncthreads();
    float a = 0.f, b = 0.f;
    for (int k = 0; k < 256; ++k) {
        float hv = hrow[k];
        a += hv * W1[(size_t)k*256 + t];
        b += hv * W1[(size_t)(256+k)*256 + t];
    }
    HA[(size_t)r*256 + t] = a;
    HB[(size_t)r*256 + t] = b;
}

// U[p] = act(HA[i]+HB[j]+b1), V[p] = act(HA[j]+HB[i]+b1) -> bf16
__global__ void assemble_uv(const float* __restrict__ HA, const float* __restrict__ HB,
                            const float* __restrict__ b1, const int* __restrict__ pairs,
                            bf16* __restrict__ U, bf16* __restrict__ V){
    int idx = blockIdx.x*blockDim.x + threadIdx.x;
    if (idx >= 32640*256) return;
    int p = idx >> 8, c = idx & 255;
    int pk = pairs[p]; int i = pk & 0xffff, j = pk >> 16;
    float bb = b1[c];
    U[idx] = (bf16)silu_g(HA[i*256+c] + HB[j*256+c] + bb);
    V[idx] = (bf16)silu_g(HA[j*256+c] + HB[i*256+c] + bb);
}

// partial products over 255-pair chunks (128*255 == 32640), fixed order
__global__ void prodpart(const float* __restrict__ OU, const float* __restrict__ OV,
                         float* __restrict__ part){
    int b = blockIdx.x, c = threadIdx.x;  // 128 blocks x 256 threads
    float pr = 1.f;
    int p0 = b * 255;
    for (int q = 0; q < 255; ++q) {
        size_t idx = (size_t)(p0 + q)*256 + c;
        pr *= (OU[idx] - OV[idx]);
    }
    part[b*256 + c] = pr;
}

__global__ void prodreduce(const float* __restrict__ part, float* __restrict__ avec){
    int c = threadIdx.x;
    float pr = 1.f;
    for (int b = 0; b < 128; ++b) pr *= part[b*256 + c];
    avec[c] = pr;
}

// net_mlp2 head for both spins + sum of log|A|
__global__ void final_kernel(const float* __restrict__ avec,
                             const float* __restrict__ W1, const float* __restrict__ b1,
                             const float* __restrict__ W2, const float* __restrict__ b2,
                             float* __restrict__ out){
    __shared__ float h64[64];
    int t = threadIdx.x;  // 64 threads
    float slog = 0.f;
    for (int spin = 0; spin < 2; ++spin) {
        const float* a = avec + spin*256;
        float acc = b1[t];
        for (int k = 0; k < 256; ++k) acc += a[k]*W1[k*64 + t];
        h64[t] = tanh_g(acc);
        __syncthreads();
        if (t == 0) {
            float o = b2[0];
            for (int k = 0; k < 64; ++k) o += h64[k]*W2[k];
            o = tanh_g(o);
            slog += logf(fabsf(o));
        }
        __syncthreads();
    }
    if (t == 0) out[0] = slog;
}

// ---------------------------------------------------------------------------

extern "C" void kernel_launch(void* const* d_in, const int* in_sizes, int n_in,
                              void* d_out, int out_size, void* d_ws, size_t ws_size,
                              hipStream_t stream)
{
    (void)in_sizes; (void)n_in; (void)out_size; (void)ws_size;

    // ---- inputs: setup_inputs() dict leaves in insertion order ----
    const float* encW = (const float*)d_in[0];   // (4,32)
    const float* encb = (const float*)d_in[1];   // (32,)
    const float* nuc  = (const float*)d_in[2];   // (32,32)
    const float* m1W  = (const float*)d_in[3];   // (32,32)
    const float* m1b  = (const float*)d_in[4];
    const float* m2W  = (const float*)d_in[5];   // (32,32)
    const float* m2b  = (const float*)d_in[6];
    const float* foW[4] = {(const float*)d_in[7],  (const float*)d_in[12], (const float*)d_in[17], (const float*)d_in[22]};
    const float* fob[4] = {(const float*)d_in[8],  (const float*)d_in[13], (const float*)d_in[18], (const float*)d_in[23]};
    const float* fgW[4] = {(const float*)d_in[9],  (const float*)d_in[14], (const float*)d_in[19], (const float*)d_in[24]};
    const float* fpW[3] = {(const float*)d_in[10], (const float*)d_in[15], (const float*)d_in[20]};
    const float* fpb[3] = {(const float*)d_in[11], (const float*)d_in[16], (const float*)d_in[21]};
    const float* n1W0 = (const float*)d_in[25];  // (512,256)
    const float* n1b0 = (const float*)d_in[26];
    const float* n1W1 = (const float*)d_in[27];  // (256,256)
    const float* n1b1 = (const float*)d_in[28];
    const float* n2W0 = (const float*)d_in[29];  // (256,64)
    const float* n2b0 = (const float*)d_in[30];
    const float* n2W1 = (const float*)d_in[31];  // (64,1)
    const float* n2b1 = (const float*)d_in[32];
    const float* el   = (const float*)d_in[33];  // (512,3)
    const float* at   = (const float*)d_in[34];  // (32,3)

    // ---- workspace bump allocator (256B aligned) ----
    char* base = (char*)d_ws; size_t off = 0;
    auto alloc = [&](size_t bytes) -> void* {
        void* p = base + off;
        off = (off + bytes + 255) & ~(size_t)255;
        return p;
    };
    float* hone0  = (float*)alloc(512*32*4);
    float* hone   = (float*)alloc(512*256*4);
    float* ht0    = (float*)alloc((size_t)512*512*4*4);
    float* ht     = (float*)alloc((size_t)512*512*32*4);
    bf16*  htbf   = (bf16*) alloc((size_t)512*512*32*2);
    float* upm    = (float*)alloc(512*32*4);
    float* dnm    = (float*)alloc(512*32*4);
    float* rowbias= (float*)alloc(256*4);
    bf16*  onein  = (bf16*) alloc((size_t)512*320*2);
    bf16*  wot[4];
    wot[0] = (bf16*)alloc(256*64*2);
    wot[1] = (bf16*)alloc(256*320*2);
    wot[2] = (bf16*)alloc(256*320*2);
    wot[3] = (bf16*)alloc(256*320*2);
    bf16*  wpt1   = (bf16*) alloc(32*32*2);
    bf16*  wpt2   = (bf16*) alloc(32*32*2);
    bf16*  w2t    = (bf16*) alloc(256*256*2);
    int*   pairs  = (int*)  alloc(32640*4);
    float* HA     = (float*)alloc(256*256*4);
    float* HB     = (float*)alloc(256*256*4);
    bf16*  U      = (bf16*) alloc((size_t)32640*256*2);
    bf16*  V      = (bf16*) alloc((size_t)32640*256*2);
    float* OU     = (float*)alloc((size_t)32640*256*4);
    float* OV     = (float*)alloc((size_t)32640*256*4);
    float* part   = (float*)alloc(128*256*4);
    float* avec   = (float*)alloc(512*4);

    auto gemm = [&](const bf16* A, const bf16* Bt, const float* bias, const float* resid,
                    float* C, int M, int K, int N, float alpha, int act){
        if ((N >> 4) % 4 == 0) {
            int waves  = (M/16) * ((N/16)/4);
            gemm_bf16_wmma<4><<<(waves + 3)/4, 128, 0, stream>>>(A, Bt, bias, resid, C, M, K, N, alpha, act);
        } else {
            int waves  = (M/16) * ((N/16)/2);
            gemm_bf16_wmma<2><<<(waves + 3)/4, 128, 0, stream>>>(A, Bt, bias, resid, C, M, K, N, alpha, act);
        }
    };

    // ---- weight prep (bf16 transposed, K zero-padded) ----
    tconv_k<<<(256*64 +255)/256, 256, 0, stream>>>(foW[0], wot[0],  40, 256,  64);
    tconv_k<<<(256*320+255)/256, 256, 0, stream>>>(foW[1], wot[1], 320, 256, 320);
    tconv_k<<<(256*320+255)/256, 256, 0, stream>>>(foW[2], wot[2], 320, 256, 320);
    tconv_k<<<(256*320+255)/256, 256, 0, stream>>>(foW[3], wot[3], 320, 256, 320);
    tconv_k<<<(32*32  +255)/256, 256, 0, stream>>>(fpW[1], wpt1,   32,  32,  32);
    tconv_k<<<(32*32  +255)/256, 256, 0, stream>>>(fpW[2], wpt2,   32,  32,  32);
    tconv_k<<<(256*256+255)/256, 256, 0, stream>>>(n1W1,   w2t,   256, 256, 256);
    pair_idx_k<<<256, 256, 0, stream>>>(pairs);

    // ---- encoding ----
    encode_kernel<<<512, 32, 0, stream>>>(encW, encb, nuc, m1W, m1b, m2W, m2b, el, at, hone0);
    htwo_init<<<(512*512+255)/256, 256, 0, stream>>>(el, ht0);

    // ---- fermi layer 0 (s=32,p=4 -> 256/32) ----
    pair_mean<<<(512*4+255)/256, 256, 0, stream>>>(ht0, 4, upm, dnm);
    gmean_rowbias<<<1, 256, 0, stream>>>(hone0, 32, fgW[0], fob[0], rowbias);
    assemble_onein<<<(512*64+255)/256, 256, 0, stream>>>(hone0, 32, upm, dnm, 4, onein, 64);
    gemm(onein, wot[0], rowbias, nullptr, hone, 512, 64, 256, RSQRT2, 1);
    pair0_kernel<<<(262144*32+255)/256, 256, 0, stream>>>(ht0, fpW[0], fpb[0], ht);

    // ---- fermi layers 1,2 (with pair WMMA) ----
    bf16* wptL[3] = {nullptr, wpt1, wpt2};
    for (int l = 1; l <= 2; ++l) {
        pair_mean<<<(512*32+255)/256, 256, 0, stream>>>(ht, 32, upm, dnm);
        gmean_rowbias<<<1, 256, 0, stream>>>(hone, 256, fgW[l], fob[l], rowbias);
        assemble_onein<<<(512*320+255)/256, 256, 0, stream>>>(hone, 256, upm, dnm, 32, onein, 320);
        gemm(onein, wot[l], rowbias, hone, hone, 512, 320, 256, RSQRT2, 1);
        f32_to_bf16_k<<<(512*512*32+255)/256, 256, 0, stream>>>(ht, htbf, 512*512*32);
        gemm(htbf, wptL[l], fpb[l], ht, ht, 262144, 32, 32, 1.0f, 1);
    }

    // ---- fermi layer 3 (no pair) ----
    pair_mean<<<(512*32+255)/256, 256, 0, stream>>>(ht, 32, upm, dnm);
    gmean_rowbias<<<1, 256, 0, stream>>>(hone, 256, fgW[3], fob[3], rowbias);
    assemble_onein<<<(512*320+255)/256, 256, 0, stream>>>(hone, 256, upm, dnm, 32, onein, 320);
    gemm(onein, wot[3], rowbias, hone, hone, 512, 320, 256, RSQRT2, 1);

    // ---- antisymmetrizer per spin ----
    for (int spin = 0; spin < 2; ++spin) {
        hab_kernel<<<256, 256, 0, stream>>>(hone + (size_t)spin*256*256, n1W0, HA, HB);
        assemble_uv<<<(32640*256+255)/256, 256, 0, stream>>>(HA, HB, n1b0, pairs, U, V);
        gemm(U, w2t, n1b1, nullptr, OU, 32640, 256, 256, 1.0f, 1);
        gemm(V, w2t, n1b1, nullptr, OV, 32640, 256, 256, 1.0f, 1);
        prodpart<<<128, 256, 0, stream>>>(OU, OV, part);
        prodreduce<<<1, 256, 0, stream>>>(part, avec + spin*256);
    }

    final_kernel<<<1, 64, 0, stream>>>(avec, n2W0, n2b0, n2W1, n2b1, (float*)d_out);
}
